// decoderLayer_44006234915478
// MI455X (gfx1250) — matrix operand order repaired and checked by reading, not compiled, not run on hardware
//
#include <hip/hip_runtime.h>
#include <hip/hip_bf16.h>

typedef __bf16 bf16_t;
typedef __attribute__((ext_vector_type(16))) __bf16 v16bf;
typedef __attribute__((ext_vector_type(8)))  float   v8f;
typedef int v4i __attribute__((vector_size(16)));

#define EDIM 1024
#define SDIM 2048
#define BDIM 8
#define NROW (BDIM * SDIM)   // 16384

#define BM 128
#define BN 128
#define BK 32
#define NT (BN / 16)         // 8 column tiles per wave stripe

#define AS1 __attribute__((address_space(1)))
#define AS3 __attribute__((address_space(3)))

// ---- CDNA5 async global->LDS copy (ASYNCcnt-tracked), with fallback --------
#if defined(__has_builtin)
#if __has_builtin(__builtin_amdgcn_global_load_async_to_lds_b128)
#define HAVE_ASYNC_LDS 1
#endif
#endif
#ifndef HAVE_ASYNC_LDS
#define HAVE_ASYNC_LDS 0
#endif

#if HAVE_ASYNC_LDS
__device__ __forceinline__ void async_cp16(const bf16_t* g, bf16_t* l) {
  // 16 bytes per lane, global -> LDS, tracked with ASYNCcnt
  __builtin_amdgcn_global_load_async_to_lds_b128(
      (AS1 v4i*)(AS1 void*)(bf16_t*)g, (AS3 v4i*)(AS3 void*)l, 0, 0);
}
__device__ __forceinline__ void wait_async_le4() {
#if __has_builtin(__builtin_amdgcn_s_wait_asynccnt)
  __builtin_amdgcn_s_wait_asynccnt(4);
#else
  asm volatile("s_wait_asynccnt 0x4" ::: "memory");
#endif
}
__device__ __forceinline__ void wait_async_le0() {
#if __has_builtin(__builtin_amdgcn_s_wait_asynccnt)
  __builtin_amdgcn_s_wait_asynccnt(0);
#else
  asm volatile("s_wait_asynccnt 0x0" ::: "memory");
#endif
}
#else
__device__ __forceinline__ void wait_async_le4() {}
__device__ __forceinline__ void wait_async_le0() {}
#endif

// ---------------------------------------------------------------------------
// fp32 -> bf16 cast (4 elems / thread)
// ---------------------------------------------------------------------------
__global__ void cast_f32_to_bf16(const float* __restrict__ in,
                                 bf16_t* __restrict__ out, int n) {
  int i = (blockIdx.x * 256 + threadIdx.x) * 4;
  if (i + 3 < n) {
    float4 v = *reinterpret_cast<const float4*>(in + i);
    out[i + 0] = (bf16_t)v.x;
    out[i + 1] = (bf16_t)v.y;
    out[i + 2] = (bf16_t)v.z;
    out[i + 3] = (bf16_t)v.w;
  }
}

// ---------------------------------------------------------------------------
// bf16 WMMA GEMM: C[M,N] = A[M,K]*B[K,N] (+bias)(ReLU), fp32 acc.
// 256 threads = 8 waves; 128x128 block tile; wave owns a 16-row stripe and
// 8 16x16 C tiles. K-step 32, double-buffered LDS with async copies.
// Requires M%128==0, N%128==0, K%32==0 (true for every call here).
// ---------------------------------------------------------------------------
__global__ __launch_bounds__(256) void gemm_bf16_wmma(
    const bf16_t* __restrict__ A, const bf16_t* __restrict__ B,
    float* __restrict__ Cf, bf16_t* __restrict__ Ch,
    const float* __restrict__ bias, int relu,
    int M, int N, int K, int lda, int ldb, int ldc) {
  __shared__ __align__(16) bf16_t sA[2][BM][BK];   // 2 x 8 KB
  __shared__ __align__(16) bf16_t sB[2][BK][BN];   // 2 x 8 KB

  const int tid  = threadIdx.x;
  const int wid  = tid >> 5;       // wave id 0..7
  const int lane = tid & 31;
  const int hl   = lane >> 4;      // lane-half
  const int l16  = lane & 15;

  const int rowBase = blockIdx.y * BM;
  const int colBase = blockIdx.x * BN;

  // staging decomposition: A = 128 rows x 2 segs of 16; B = 32 rows x 8 segs of 16
  const int arow = tid >> 1, aseg = tid & 1;
  const int brow = tid >> 3, bseg = tid & 7;
  const bf16_t* gA = A + (size_t)(rowBase + arow) * lda + aseg * 16;
  const bf16_t* gB = B + (size_t)brow * ldb + colBase + bseg * 16;

  v8f acc[NT];
#pragma unroll
  for (int t = 0; t < NT; ++t)
#pragma unroll
    for (int r = 0; r < 8; ++r) acc[t][r] = 0.0f;

  auto stage = [&](int buf, int k0) {
#if HAVE_ASYNC_LDS
    async_cp16(gA + k0,                   &sA[buf][arow][aseg * 16]);
    async_cp16(gA + k0 + 8,               &sA[buf][arow][aseg * 16 + 8]);
    async_cp16(gB + (size_t)k0 * ldb,     &sB[buf][brow][bseg * 16]);
    async_cp16(gB + (size_t)k0 * ldb + 8, &sB[buf][brow][bseg * 16 + 8]);
#else
    const uint4* a0 = reinterpret_cast<const uint4*>(gA + k0);
    uint4* da = reinterpret_cast<uint4*>(&sA[buf][arow][aseg * 16]);
    da[0] = a0[0];
    da[1] = a0[1];
    const uint4* b0 = reinterpret_cast<const uint4*>(gB + (size_t)k0 * ldb);
    uint4* db = reinterpret_cast<uint4*>(&sB[buf][brow][bseg * 16]);
    db[0] = b0[0];
    db[1] = b0[1];
#endif
  };

  auto compute = [&](int cur) {
    // A fragment (ISA 7.12.2, 16-bit A 16x32):
    // lanes 0-15: K = 0..7 / 16..23 ; lanes 16-31: K = 8..15 / 24..31
    v16bf afrag;
#pragma unroll
    for (int v = 0; v < 8; ++v) {
      const int kb = (v < 4 ? 2 * v : 16 + 2 * (v - 4)) + 8 * hl;
      afrag[2 * v]     = sA[cur][wid * 16 + l16][kb];
      afrag[2 * v + 1] = sA[cur][wid * 16 + l16][kb + 1];
    }
    // B fragments: lane holds column N=l16, K in same order as A fragment
#pragma unroll
    for (int t = 0; t < NT; ++t) {
      v16bf bfrag;
#pragma unroll
      for (int e = 0; e < 16; ++e) bfrag[e] = sB[cur][(e < 8 ? e : 8 + e) + 8 * hl][t * 16 + l16];
      acc[t] = __builtin_amdgcn_wmma_f32_16x16x32_bf16(
          false, afrag, false, bfrag, (short)0, acc[t], false, false);
    }
  };

  const int nk = K / BK;
  stage(0, 0);
  // steady state: next tile in flight while current tile is consumed
  for (int kt = 0; kt < nk - 1; ++kt) {
    stage((kt + 1) & 1, (kt + 1) * BK);
    wait_async_le4();   // in-order retirement => current tile resident
    __syncthreads();
    compute(kt & 1);
    __syncthreads();    // protect current buf before it is restaged
  }
  // drain: last tile
  wait_async_le0();
  __syncthreads();
  compute((nk - 1) & 1);

  // epilogue: C layout m = r + 8*hl, n = l16
#pragma unroll
  for (int t = 0; t < NT; ++t) {
    const int col = colBase + t * 16 + l16;
    const float bv = bias ? bias[col] : 0.0f;
#pragma unroll
    for (int r = 0; r < 8; ++r) {
      const int row = rowBase + wid * 16 + r + 8 * hl;
      float v = acc[t][r] + bv;
      if (relu) v = fmaxf(v, 0.0f);
      const size_t idx = (size_t)row * ldc + col;
      if (Cf) Cf[idx] = v;
      if (Ch) Ch[idx] = (bf16_t)v;
    }
  }
}

// ---------------------------------------------------------------------------
// bf16 tiled transpose: out[c][r] = in[r][c]; rows%32==0, cols%32==0
// ---------------------------------------------------------------------------
__global__ void transpose_bf16(const bf16_t* __restrict__ in,
                               bf16_t* __restrict__ out, int rows, int cols) {
  __shared__ bf16_t tile[32][33];
  const int c0 = blockIdx.x * 32, r0 = blockIdx.y * 32;
  const int tx = threadIdx.x, ty = threadIdx.y;
#pragma unroll
  for (int j = 0; j < 32; j += 8)
    tile[ty + j][tx] = in[(size_t)(r0 + ty + j) * cols + c0 + tx];
  __syncthreads();
#pragma unroll
  for (int j = 0; j < 32; j += 8)
    out[(size_t)(c0 + ty + j) * rows + r0 + tx] = tile[tx][ty + j];
}

// ---------------------------------------------------------------------------
// lengths[b] heuristic: count nonzeros in energy row 0 (pre-mask, pre-scale)
// ---------------------------------------------------------------------------
__global__ void lengths_kernel(const float* __restrict__ energy0, int S,
                               int* __restrict__ lenOut) {
  __shared__ int sred[256];
  int cnt = 0;
  for (int k = threadIdx.x; k < S; k += 256) cnt += (energy0[k] != 0.0f);
  sred[threadIdx.x] = cnt;
  __syncthreads();
  for (int o = 128; o > 0; o >>= 1) {
    if (threadIdx.x < o) sred[threadIdx.x] += sred[threadIdx.x + o];
    __syncthreads();
  }
  if (threadIdx.x == 0) *lenOut = sred[0];
}

// ---------------------------------------------------------------------------
// masked, scaled row softmax: energy[q,:] (fp32) -> P[q,:] (bf16). S == 2048.
// ---------------------------------------------------------------------------
__global__ __launch_bounds__(256) void softmax_kernel(
    const float* __restrict__ energy, bf16_t* __restrict__ P,
    const int* __restrict__ lenPtr, int S, int causal, float scale) {
  const int q   = blockIdx.x;
  const int tid = threadIdx.x;
  const int len = *lenPtr;
  const float* row = energy + (size_t)q * S;
  bf16_t* prow = P + (size_t)q * S;
  __shared__ float sred[256];

  float vals[8];
  float m = -INFINITY;
#pragma unroll
  for (int i = 0; i < 8; ++i) {
    const int k = tid + i * 256;
    float v = row[k] * scale;
    const bool mask = (k >= len) || (causal && (k > q));
    v = mask ? -INFINITY : v;
    vals[i] = v;
    m = fmaxf(m, v);
  }
  sred[tid] = m;
  __syncthreads();
  for (int o = 128; o > 0; o >>= 1) {
    if (tid < o) sred[tid] = fmaxf(sred[tid], sred[tid + o]);
    __syncthreads();
  }
  m = sred[0];
  __syncthreads();

  float s = 0.0f;
#pragma unroll
  for (int i = 0; i < 8; ++i) {
    const float e = __expf(vals[i] - m);
    vals[i] = e;
    s += e;
  }
  sred[tid] = s;
  __syncthreads();
  for (int o = 128; o > 0; o >>= 1) {
    if (tid < o) sred[tid] += sred[tid + o];
    __syncthreads();
  }
  const float inv = 1.0f / sred[0];
#pragma unroll
  for (int i = 0; i < 8; ++i) prow[tid + i * 256] = (bf16_t)(vals[i] * inv);
}

// ---------------------------------------------------------------------------
// y = LayerNorm(x + z) per row of 1024; writes fp32 and optional bf16 copy
// ---------------------------------------------------------------------------
__global__ __launch_bounds__(256) void add_ln_kernel(
    const float* __restrict__ x, const float* __restrict__ z,
    const float* __restrict__ g, const float* __restrict__ b,
    float* __restrict__ yf, bf16_t* __restrict__ yb) {
  const int row = blockIdx.x;
  const int tid = threadIdx.x;
  const float* xr = x + (size_t)row * EDIM;
  const float* zr = z + (size_t)row * EDIM;
  __shared__ float sred[256];

  float loc[4];
  float s = 0.0f;
#pragma unroll
  for (int i = 0; i < 4; ++i) {
    const int e = tid + i * 256;
    const float v = xr[e] + zr[e];
    loc[i] = v;
    s += v;
  }
  sred[tid] = s;
  __syncthreads();
  for (int o = 128; o > 0; o >>= 1) {
    if (tid < o) sred[tid] += sred[tid + o];
    __syncthreads();
  }
  const float mu = sred[0] * (1.0f / (float)EDIM);
  __syncthreads();

  float s2 = 0.0f;
#pragma unroll
  for (int i = 0; i < 4; ++i) {
    const float d = loc[i] - mu;
    s2 += d * d;
  }
  sred[tid] = s2;
  __syncthreads();
  for (int o = 128; o > 0; o >>= 1) {
    if (tid < o) sred[tid] += sred[tid + o];
    __syncthreads();
  }
  const float rs = rsqrtf(sred[0] * (1.0f / (float)EDIM) + 1e-5f);

#pragma unroll
  for (int i = 0; i < 4; ++i) {
    const int e = tid + i * 256;
    const float y = (loc[i] - mu) * rs * g[e] + b[e];
    const size_t idx = (size_t)row * EDIM + e;
    if (yf) yf[idx] = y;
    if (yb) yb[idx] = (bf16_t)y;
  }
}

// ---------------------------------------------------------------------------
// host orchestration
// ---------------------------------------------------------------------------
extern "C" void kernel_launch(void* const* d_in, const int* in_sizes, int n_in,
                              void* d_out, int out_size, void* d_ws,
                              size_t ws_size, hipStream_t stream) {
  (void)in_sizes; (void)n_in; (void)out_size; (void)ws_size;
  const float* x     = (const float*)d_in[0];
  const float* ctx   = (const float*)d_in[1];
  const float* fc1_b = (const float*)d_in[9];
  const float* fc2_b = (const float*)d_in[11];
  const float* ln1_g = (const float*)d_in[12];
  const float* ln1_b = (const float*)d_in[13];
  const float* ln2_g = (const float*)d_in[14];
  const float* ln2_b = (const float*)d_in[15];
  const float* ln3_g = (const float*)d_in[16];
  const float* ln3_b = (const float*)d_in[17];

  char* p = (char*)d_ws;
  auto carve = [&](size_t bytes) -> void* {
    void* r = (void*)p;
    p += (bytes + 255) & ~(size_t)255;
    return r;
  };
  const size_t actE = (size_t)NROW * EDIM;
  bf16_t* xb  = (bf16_t*)carve(actE * 2);
  bf16_t* cb  = (bf16_t*)carve(actE * 2);
  bf16_t* wb  = (bf16_t*)carve((size_t)8 * EDIM * EDIM * 2);
  bf16_t* Qb  = (bf16_t*)carve(actE * 2);
  bf16_t* Kb  = (bf16_t*)carve(actE * 2);
  bf16_t* Vb  = (bf16_t*)carve(actE * 2);
  bf16_t* KT  = (bf16_t*)carve(actE * 2);
  float*  eng = (float*)carve((size_t)SDIM * SDIM * 4);   // per-batch reuse
  bf16_t* Pb  = (bf16_t*)carve((size_t)SDIM * SDIM * 2);  // per-batch reuse
  float*  zf  = (float*)carve(actE * 4);
  float*  x1f = (float*)carve(actE * 4);
  bf16_t* x1b = (bf16_t*)carve(actE * 2);
  float*  x2f = (float*)carve(actE * 4);
  int*    len = (int*)carve(256);
  bf16_t* x2b = xb;  // x bf16 dead after self-attn projections
  bf16_t* h1b = Qb;  // Q bf16 dead after attention

  const size_t EE = (size_t)EDIM * EDIM;
  const int wIdx[8] = {2, 3, 4, 5, 6, 7, 8, 10};
  for (int i = 0; i < 8; ++i)
    cast_f32_to_bf16<<<(int)(EE / 1024), 256, 0, stream>>>(
        (const float*)d_in[wIdx[i]], wb + (size_t)i * EE, (int)EE);
  cast_f32_to_bf16<<<(int)(actE / 1024), 256, 0, stream>>>(x, xb, (int)actE);
  cast_f32_to_bf16<<<(int)(actE / 1024), 256, 0, stream>>>(ctx, cb, (int)actE);

  auto gemm = [&](const bf16_t* A, const bf16_t* Bm, float* Cf, bf16_t* Ch,
                  const float* bias, int relu, int M, int N, int K, int lda,
                  int ldb, int ldc) {
    dim3 grid(N / BN, M / BM);
    gemm_bf16_wmma<<<grid, 256, 0, stream>>>(A, Bm, Cf, Ch, bias, relu, M, N,
                                             K, lda, ldb, ldc);
  };

  auto attention = [&](const bf16_t* Qs, const bf16_t* Ks, const bf16_t* Vs,
                       int causal, float* zout) {
    for (int b = 0; b < BDIM; ++b)
      transpose_bf16<<<dim3(EDIM / 32, SDIM / 32), dim3(32, 8), 0, stream>>>(
          Ks + (size_t)b * SDIM * EDIM, KT + (size_t)b * EDIM * SDIM, SDIM, EDIM);
    for (int b = 0; b < BDIM; ++b) {
      gemm(Qs + (size_t)b * SDIM * EDIM, KT + (size_t)b * EDIM * SDIM, eng,
           nullptr, nullptr, 0, SDIM, SDIM, EDIM, EDIM, SDIM, SDIM);
      lengths_kernel<<<1, 256, 0, stream>>>(eng, SDIM, len);
      softmax_kernel<<<SDIM, 256, 0, stream>>>(eng, Pb, len, SDIM, causal,
                                               0.125f);  // 1/sqrt(64)
      gemm(Pb, Vs + (size_t)b * SDIM * EDIM, zout + (size_t)b * SDIM * EDIM,
           nullptr, nullptr, 0, SDIM, EDIM, SDIM, SDIM, EDIM, EDIM);
    }
  };

  // ---- self-attention + LN1 ----
  gemm(xb, wb + 0 * EE, nullptr, Qb, nullptr, 0, NROW, EDIM, EDIM, EDIM, EDIM, EDIM);
  gemm(xb, wb + 1 * EE, nullptr, Kb, nullptr, 0, NROW, EDIM, EDIM, EDIM, EDIM, EDIM);
  gemm(xb, wb + 2 * EE, nullptr, Vb, nullptr, 0, NROW, EDIM, EDIM, EDIM, EDIM, EDIM);
  attention(Qb, Kb, Vb, /*causal=*/1, zf);
  add_ln_kernel<<<NROW, 256, 0, stream>>>(x, zf, ln1_g, ln1_b, x1f, x1b);

  // ---- cross-attention + LN2 ----
  gemm(x1b, wb + 3 * EE, nullptr, Qb, nullptr, 0, NROW, EDIM, EDIM, EDIM, EDIM, EDIM);
  gemm(cb,  wb + 4 * EE, nullptr, Kb, nullptr, 0, NROW, EDIM, EDIM, EDIM, EDIM, EDIM);
  gemm(cb,  wb + 5 * EE, nullptr, Vb, nullptr, 0, NROW, EDIM, EDIM, EDIM, EDIM, EDIM);
  attention(Qb, Kb, Vb, /*causal=*/0, zf);
  add_ln_kernel<<<NROW, 256, 0, stream>>>(x1f, zf, ln2_g, ln2_b, x2f, x2b);

  // ---- FFN + LN3 ----
  gemm(x2b, wb + 6 * EE, nullptr, h1b, fc1_b, /*relu=*/1, NROW, EDIM, EDIM, EDIM, EDIM, EDIM);
  gemm(h1b, wb + 7 * EE, zf, nullptr, fc2_b, /*relu=*/0, NROW, EDIM, EDIM, EDIM, EDIM, EDIM);
  add_ln_kernel<<<NROW, 256, 0, stream>>>(x2f, zf, ln3_g, ln3_b, (float*)d_out,
                                          nullptr);
}
